// MHSA_21328807592167
// MI455X (gfx1250) — matrix operand order repaired and checked by reading, not compiled, not run on hardware
//
#include <hip/hip_runtime.h>
#include <hip/hip_bf16.h>

// ---------------------------------------------------------------------------
// MHSA with relative-position bias, MI455X (gfx1250, wave32, WMMA bf16).
// energy[m,n] = q_m.k_n + pos_m.q_n  -> softmax(n) -> out[:,m] = V attn[m,:]^T
// Register tokens (reg_qk/reg_v) are dead code in the reference (group 4 is
// dropped by out[:, :head]) and are skipped.
// Attention uses flash-style online softmax; K/Q-key/V blocks are staged into
// LDS with GLOBAL_LOAD_ASYNC_TO_LDS_B128 (double-buffered, ASYNCcnt).
// ---------------------------------------------------------------------------

typedef __attribute__((ext_vector_type(16))) __bf16 v16bf;
typedef __attribute__((ext_vector_type(8)))  __bf16 bf16x8;
typedef __attribute__((ext_vector_type(8)))  float  v8f;
typedef __attribute__((ext_vector_type(4)))  float  f32x4;

#define BB    16
#define CCH   256
#define HEADS 4
#define DD    64
#define NNSP  1024

__device__ __forceinline__ __bf16 f2bf(float f) { return (__bf16)f; }

__device__ __forceinline__ v16bf make_frag(bf16x8 c0, bf16x8 c1) {
  v16bf a;
#pragma unroll
  for (int i = 0; i < 8; ++i) { a[i] = c0[i]; a[i + 8] = c1[i]; }
  return a;
}

// load 8 consecutive f32 and convert to bf16x8
__device__ __forceinline__ bf16x8 cvt8(const float* __restrict__ p) {
  f32x4 a = *(const f32x4*)p;
  f32x4 b = *(const f32x4*)(p + 4);
  bf16x8 r;
#pragma unroll
  for (int i = 0; i < 4; ++i) { r[i] = (__bf16)a[i]; r[i + 4] = (__bf16)b[i]; }
  return r;
}

// Async 16B copy global -> LDS (per lane). lds_off is a byte offset into the
// workgroup LDS allocation (dynamic LDS starts at 0 when no static LDS).
__device__ __forceinline__ void async_cp16(unsigned lds_off, const void* g) {
  unsigned long long ga = (unsigned long long)g;
  asm volatile("global_load_async_to_lds_b128 %0, %1, off"
               :: "v"(lds_off), "v"(ga) : "memory");
}
__device__ __forceinline__ void wait_async0() {
  asm volatile("s_wait_asynccnt 0x0" ::: "memory");
}

// ---------------------------------------------------------------------------
// pos[h, n, d] = rel_h[h,d,n%32] + rel_w[h,d,n/32]   (bf16)
// ---------------------------------------------------------------------------
__global__ __launch_bounds__(256) void pos_kernel(
    const float* __restrict__ rel_h, const float* __restrict__ rel_w,
    __bf16* __restrict__ Pt) {
  int idx = blockIdx.x * 256 + threadIdx.x;     // (h*NN + n)*DD + d
  int d = idx & (DD - 1);
  int hn = idx / DD;
  int n = hn & (NNSP - 1);
  int h = hn / NNSP;
  float v = rel_h[(h * DD + d) * 32 + (n & 31)] +
            rel_w[(h * DD + d) * 32 + (n >> 5)];
  Pt[idx] = f2bf(v);
}

// ---------------------------------------------------------------------------
// QKV projection. Grid: B * (N/32) blocks x 256 threads (8 waves).
//   Qt,Kt : [B,H,N,64] bf16 (d-contiguous, for A/B score fragments)
//   Vb    : [B,H,64,N] bf16 (n-contiguous, for PV A fragments)
// ---------------------------------------------------------------------------
__global__ __launch_bounds__(256) void proj_kernel(
    const float* __restrict__ x,
    const float* __restrict__ Wq, const float* __restrict__ bq,
    const float* __restrict__ Wk, const float* __restrict__ bk,
    const float* __restrict__ Wv, const float* __restrict__ bv,
    __bf16* __restrict__ Qt, __bf16* __restrict__ Kt, __bf16* __restrict__ Vb) {
  __shared__ __bf16 xT[32 * 264];   // [n_local][c], pitch 264 (bank-safe, 16B rows)

  const int tid = threadIdx.x;
  const int b  = blockIdx.x >> 5;
  const int n0 = (blockIdx.x & 31) * 32;

  { // cooperative transpose-load of x[b, :, n0:n0+32] -> LDS bf16
    const int nn = tid & 31;
    const int cb = tid >> 5;  // 0..7
    const float* xb = x + (size_t)b * CCH * NNSP + n0 + nn;
#pragma unroll 4
    for (int it = 0; it < 32; ++it) {
      int c = it * 8 + cb;
      xT[nn * 264 + c] = f2bf(xb[(size_t)c * NNSP]);
    }
  }
  __syncthreads();

  const int wid  = tid >> 5;
  const int L    = tid & 31;
  const int l16  = L & 15;
  const int half = L >> 4;

  // 96 tile jobs: proj(3) x otile(16) x ntile(2); 12 per wave
  for (int t = 0; t < 12; ++t) {
    int j    = wid + 8 * t;
    int proj = j >> 5;        // 0=Q 1=K 2=V
    int rem  = j & 31;
    int ot   = rem >> 1;
    int nt   = rem & 1;
    const float* Wm   = (proj == 0) ? Wq : (proj == 1) ? Wk : Wv;
    const float* bias = (proj == 0) ? bq : (proj == 1) ? bk : bv;

    v8f acc;
#pragma unroll
    for (int r = 0; r < 8; ++r) acc[r] = 0.0f;

    if (proj < 2) {
      // A = xT rows (M=n), B = W^T (N=o): C rows = n, cols = o
      const int nrow = nt * 16 + l16;
      const int ocol = ot * 16 + l16;
      __builtin_prefetch(Wm + (size_t)ocol * CCH, 0, 1);
#pragma unroll
      for (int kk = 0; kk < 8; ++kk) {
        int c0 = kk * 32;
        bf16x8 a0 = *(const bf16x8*)&xT[nrow * 264 + c0 + 8 * half];
        bf16x8 a1 = *(const bf16x8*)&xT[nrow * 264 + c0 + 16 + 8 * half];
        v16bf  A  = make_frag(a0, a1);
        bf16x8 b0 = cvt8(Wm + (size_t)ocol * CCH + c0 + 8 * half);
        bf16x8 b1 = cvt8(Wm + (size_t)ocol * CCH + c0 + 16 + 8 * half);
        v16bf  Bf = make_frag(b0, b1);
        acc = __builtin_amdgcn_wmma_f32_16x16x32_bf16(false, A, false, Bf,
                                                      (short)0, acc, false, false);
      }
      float bl = bias[ocol];
      int hh = ocol >> 6, d = ocol & 63;
      __bf16* dst = (proj == 0) ? Qt : Kt;
#pragma unroll
      for (int r = 0; r < 8; ++r) {
        int n = n0 + nt * 16 + r + 8 * half;
        dst[((size_t)(b * HEADS + hh) * NNSP + n) * DD + d] = f2bf(acc[r] + bl);
      }
    } else {
      // A = W rows (M=o), B = xT cols (N=n): C rows = o, cols = n
      const int orow = ot * 16 + l16;
      const int ncol = nt * 16 + l16;
      __builtin_prefetch(Wm + (size_t)orow * CCH, 0, 1);
#pragma unroll
      for (int kk = 0; kk < 8; ++kk) {
        int c0 = kk * 32;
        bf16x8 a0 = cvt8(Wm + (size_t)orow * CCH + c0 + 8 * half);
        bf16x8 a1 = cvt8(Wm + (size_t)orow * CCH + c0 + 16 + 8 * half);
        v16bf  A  = make_frag(a0, a1);
        bf16x8 b0 = *(const bf16x8*)&xT[ncol * 264 + c0 + 8 * half];
        bf16x8 b1 = *(const bf16x8*)&xT[ncol * 264 + c0 + 16 + 8 * half];
        v16bf  Bf = make_frag(b0, b1);
        acc = __builtin_amdgcn_wmma_f32_16x16x32_bf16(false, A, false, Bf,
                                                      (short)0, acc, false, false);
      }
#pragma unroll
      for (int r = 0; r < 8; ++r) {
        int o = ot * 16 + r + 8 * half;
        int hh = o >> 6, d = o & 63;
        Vb[((size_t)(b * HEADS + hh) * DD + d) * NNSP + n0 + ncol] =
            f2bf(acc[r] + bias[o]);
      }
    }
  }
}

// ---------------------------------------------------------------------------
// Flash attention. Grid: B*H*(N/128) blocks x 256 threads (8 waves).
// Each wave owns 16 queries; workgroup stages each 32-key block (K tile,
// Q-as-key tile, V tile) into double-buffered LDS via async DMA.
//
// Dynamic-LDS byte layout (no static LDS in this kernel => base offset 0):
//   [0,     9216)  kbuf[2]   : 32 rows x pitch 72 bf16 (144 B)  = 4608 B each
//   [9216, 18432)  qbuf[2]   : same shape
//   [18432,28672)  vbuf[2]   : 64 rows x pitch 40 bf16 (80 B)   = 5120 B each
//   [28672,38912)  plds[8]   : per-wave 16 x pitch 40 bf16      = 1280 B each
//   [38912,39424)  slds[8]   : per-wave 16 f32                  = 64  B each
// ---------------------------------------------------------------------------
#define KOFF 0u
#define QOFF 9216u
#define VOFF 18432u
#define POFF 28672u
#define SOFF 38912u
#define ATTN_LDS_BYTES 39424u

__device__ __forceinline__ void stage_block(const __bf16* Kb, const __bf16* Qb,
                                            const __bf16* Vv, int n0, int tid,
                                            unsigned koff, unsigned qoff,
                                            unsigned voff) {
  // K/Q tiles: 32 rows x 128 B -> 256 x 16B chunks, 1 per thread
  int row = tid >> 3, col = tid & 7;
  async_cp16(koff + row * 144 + col * 16,
             (const char*)Kb + (size_t)(n0 + row) * 128 + col * 16);
  async_cp16(qoff + row * 144 + col * 16,
             (const char*)Qb + (size_t)(n0 + row) * 128 + col * 16);
  // V tile: 64 rows x 64 B -> 256 x 16B chunks, 1 per thread
  int vrow = tid >> 2, vcol = tid & 3;
  async_cp16(voff + vrow * 80 + vcol * 16,
             (const char*)Vv + (size_t)vrow * 2048 + n0 * 2 + vcol * 16);
}

__global__ __launch_bounds__(256) void attn_kernel(
    const __bf16* __restrict__ Qt, const __bf16* __restrict__ Kt,
    const __bf16* __restrict__ Vb, const __bf16* __restrict__ Pt,
    float* __restrict__ out) {
  extern __shared__ char smem[];

  const int tid  = threadIdx.x;
  const int wid  = tid >> 5;
  const int L    = tid & 31;
  const int l16  = L & 15;
  const int half = L >> 4;
  const int bh   = blockIdx.x >> 3;
  const int b    = bh >> 2, h = bh & 3;
  const int m0   = (blockIdx.x & 7) * 128 + wid * 16;

  const __bf16* Qb = Qt + (size_t)bh * NNSP * DD;
  const __bf16* Kb = Kt + (size_t)bh * NNSP * DD;
  const __bf16* Vv = Vb + (size_t)bh * DD * NNSP;
  const __bf16* Pb = Pt + (size_t)h * NNSP * DD;

  __bf16* pl = (__bf16*)(smem + POFF + wid * 1280);
  float*  sl = (float*)(smem + SOFF + wid * 64);

  // A fragments: queries m = m0 + l16 (q and pos, d split into 2 x K=32)
  v16bf aq[2], ap[2];
  {
    const __bf16* qr = Qb + (size_t)(m0 + l16) * DD;
    const __bf16* pr = Pb + (size_t)(m0 + l16) * DD;
#pragma unroll
    for (int f = 0; f < 2; ++f) {
      bf16x8 c0 = *(const bf16x8*)(qr + 32 * f + 8 * half);
      bf16x8 c1 = *(const bf16x8*)(qr + 32 * f + 16 + 8 * half);
      aq[f] = make_frag(c0, c1);
      bf16x8 d0 = *(const bf16x8*)(pr + 32 * f + 8 * half);
      bf16x8 d1 = *(const bf16x8*)(pr + 32 * f + 16 + 8 * half);
      ap[f] = make_frag(d0, d1);
    }
  }

  v8f O[4];
#pragma unroll
  for (int td = 0; td < 4; ++td)
#pragma unroll
    for (int r = 0; r < 8; ++r) O[td][r] = 0.0f;

  float mrun[8], lrun[8];
#pragma unroll
  for (int r = 0; r < 8; ++r) { mrun[r] = -3.0e38f; lrun[r] = 0.0f; }

  // prime buffer 0
  stage_block(Kb, Qb, Vv, 0, tid, KOFF, QOFF, VOFF);

  for (int blk = 0; blk < NNSP / 32; ++blk) {
    const int cur = blk & 1;
    wait_async0();        // this wave's DMA chunks have landed in LDS
    __syncthreads();      // everyone's chunks visible; prior reads of nxt done

    if (blk + 1 < NNSP / 32) {
      int nxt = cur ^ 1;
      stage_block(Kb, Qb, Vv, (blk + 1) * 32, tid,
                  KOFF + nxt * 4608, QOFF + nxt * 4608, VOFF + nxt * 5120);
    }

    const __bf16* kb = (const __bf16*)(smem + KOFF + cur * 4608);
    const __bf16* qb = (const __bf16*)(smem + QOFF + cur * 4608);
    const __bf16* vb = (const __bf16*)(smem + VOFF + cur * 5120);

    // ---- scores: S[m, 32 keys] via 2 subtiles x 4 WMMA (K=128 total) ----
    v8f S0, S1;
#pragma unroll
    for (int sub = 0; sub < 2; ++sub) {
      const int nl = sub * 16 + l16;   // local key row in staged tile
      const __bf16* kr = kb + nl * 72;
      const __bf16* qr = qb + nl * 72;
      v8f s;
#pragma unroll
      for (int r = 0; r < 8; ++r) s[r] = 0.0f;
#pragma unroll
      for (int f = 0; f < 2; ++f) {
        bf16x8 k0 = *(const bf16x8*)(kr + 32 * f + 8 * half);
        bf16x8 k1 = *(const bf16x8*)(kr + 32 * f + 16 + 8 * half);
        v16bf  BK = make_frag(k0, k1);
        s = __builtin_amdgcn_wmma_f32_16x16x32_bf16(false, aq[f], false, BK,
                                                    (short)0, s, false, false);
        bf16x8 q0 = *(const bf16x8*)(qr + 32 * f + 8 * half);
        bf16x8 q1 = *(const bf16x8*)(qr + 32 * f + 16 + 8 * half);
        v16bf  BQ = make_frag(q0, q1);
        s = __builtin_amdgcn_wmma_f32_16x16x32_bf16(false, ap[f], false, BQ,
                                                    (short)0, s, false, false);
      }
      if (sub == 0) S0 = s; else S1 = s;
    }

    // ---- online softmax (rows m = r + 8*half; reduce across 16 lanes) ----
    float p0[8], p1[8], alpha[8];
#pragma unroll
    for (int r = 0; r < 8; ++r) {
      float mx = fmaxf(S0[r], S1[r]);
#pragma unroll
      for (int msk = 1; msk < 16; msk <<= 1) mx = fmaxf(mx, __shfl_xor(mx, msk));
      float mnew = fmaxf(mrun[r], mx);
      alpha[r] = __expf(mrun[r] - mnew);
      p0[r] = __expf(S0[r] - mnew);
      p1[r] = __expf(S1[r] - mnew);
      float sm = p0[r] + p1[r];
#pragma unroll
      for (int msk = 1; msk < 16; msk <<= 1) sm += __shfl_xor(sm, msk);
      lrun[r] = lrun[r] * alpha[r] + sm;
      mrun[r] = mnew;
    }

    // ---- stage P (bf16) + alpha through LDS (C-layout -> B-frag layout) ----
#pragma unroll
    for (int r = 0; r < 8; ++r) {
      pl[(r + 8 * half) * 40 + l16]      = f2bf(p0[r]);
      pl[(r + 8 * half) * 40 + 16 + l16] = f2bf(p1[r]);
    }
    if (l16 == 0) {
#pragma unroll
      for (int r = 0; r < 8; ++r) sl[r + 8 * half] = alpha[r];
    }
    __syncthreads();

    float al = sl[l16];
    bf16x8 pp0 = *(const bf16x8*)&pl[l16 * 40 + 8 * half];
    bf16x8 pp1 = *(const bf16x8*)&pl[l16 * 40 + 16 + 8 * half];
    v16bf  BP  = make_frag(pp0, pp1);

    // ---- O^T[d, m] += V[d, n-block] * P^T : 4 d-tiles x 1 WMMA ----
#pragma unroll
    for (int td = 0; td < 4; ++td) {
#pragma unroll
      for (int r = 0; r < 8; ++r) O[td][r] *= al;
      const __bf16* vr = vb + (td * 16 + l16) * 40;
      bf16x8 v0 = *(const bf16x8*)(vr + 8 * half);
      bf16x8 v1 = *(const bf16x8*)(vr + 16 + 8 * half);
      v16bf  AV = make_frag(v0, v1);
      O[td] = __builtin_amdgcn_wmma_f32_16x16x32_bf16(false, AV, false, BP,
                                                      (short)0, O[td], false, false);
    }
    __syncthreads();
  }

  // ---- finalize: O / l, store f32 (m-contiguous) ----
  if (l16 == 0) {
#pragma unroll
    for (int r = 0; r < 8; ++r) sl[r + 8 * half] = lrun[r];
  }
  __syncthreads();
  float linv = 1.0f / sl[l16];
#pragma unroll
  for (int td = 0; td < 4; ++td) {
#pragma unroll
    for (int r = 0; r < 8; ++r) {
      int d = td * 16 + r + 8 * half;
      out[((size_t)(b * CCH + h * DD + d)) * NNSP + m0 + l16] = O[td][r] * linv;
    }
  }
}

// ---------------------------------------------------------------------------
extern "C" void kernel_launch(void* const* d_in, const int* in_sizes, int n_in,
                              void* d_out, int out_size, void* d_ws, size_t ws_size,
                              hipStream_t stream) {
  const float* x     = (const float*)d_in[0];
  const float* Wq    = (const float*)d_in[1];
  const float* bq    = (const float*)d_in[2];
  const float* Wk    = (const float*)d_in[3];
  const float* bk    = (const float*)d_in[4];
  const float* Wv    = (const float*)d_in[5];
  const float* bv    = (const float*)d_in[6];
  const float* rel_h = (const float*)d_in[7];
  const float* rel_w = (const float*)d_in[8];
  // d_in[9] (reg_qk) and d_in[10] (reg_v) are dead in the reference output.

  const size_t qkv_elems = (size_t)BB * HEADS * NNSP * DD;  // 4M bf16 each
  __bf16* Qt = (__bf16*)d_ws;
  __bf16* Kt = Qt + qkv_elems;
  __bf16* Vb = Kt + qkv_elems;
  __bf16* Pt = Vb + qkv_elems;  // + HEADS*NNSP*DD bf16

  pos_kernel<<<(HEADS * NNSP * DD) / 256, 256, 0, stream>>>(rel_h, rel_w, Pt);
  proj_kernel<<<BB * (NNSP / 32), 256, 0, stream>>>(x, Wq, bq, Wk, bk, Wv, bv,
                                                    Qt, Kt, Vb);
  attn_kernel<<<BB * HEADS * (NNSP / 128), 256, ATTN_LDS_BYTES, stream>>>(
      Qt, Kt, Vb, Pt, (float*)d_out);
}
